// GraphSAGE_51917564674345
// MI455X (gfx1250) — compile-verified, matrix operand order checked by reading, not run on hardware
//
#include <hip/hip_runtime.h>
#include <cmath>

// ---------------------------------------------------------------------------
// GraphSAGE forward for MI455X (gfx1250, wave32, WMMA f16->f32).
// Dominant op: adj[10000x10000] @ h[10000x128] twice -> HBM bound (800MB adj
// @ 23.3TB/s ~= 34us). f16 WMMA keeps the matrix engine off the critical path;
// adj row-normalization is fused into the SpMM via an extra ones-column WMMA
// accumulator (rowsum computed in-pass). B operand of the SpMM is a f16
// transposed copy of h so the hot loop is pure b128 loads + wmma, no branches.
// K-loop double-buffers A and preloads B tiles in groups of 4 so WMMA sources
// are not immediately overwritten (avoids WMMA->VALU hazard v_nops).
// ---------------------------------------------------------------------------

typedef __attribute__((ext_vector_type(16))) _Float16 v16h;
typedef __attribute__((ext_vector_type(8)))  _Float16 v8h;
typedef __attribute__((ext_vector_type(8)))  float    v8f;

#define NNODES 10000
#define NFEATC 2000
#define NFEC   256
#define NHC    128
#define NHEC   64
#define NOUTC  20

// ----- tile loaders ----------------------------------------------------------
// 16-bit A-matrix 16x32 layout (ISA 7.12.2): lane<16 holds row M=lane,
// halves 0..7 = K off+0..7, halves 8..15 = K 16+off+0..7, off=(lane>=16)?8:0.
// B (32x16) mirrored: lane&15 = column N, same K slot mapping.
// Pointers passed here already include the per-lane row base AND +off.

__device__ __forceinline__ v16h cvt_load32(const float* __restrict__ p) {
  float4 c0 = *(const float4*)(p);
  float4 c1 = *(const float4*)(p + 4);
  float4 c2 = *(const float4*)(p + 16);
  float4 c3 = *(const float4*)(p + 20);
  v16h o;
  o[0]=(_Float16)c0.x;  o[1]=(_Float16)c0.y;  o[2]=(_Float16)c0.z;  o[3]=(_Float16)c0.w;
  o[4]=(_Float16)c1.x;  o[5]=(_Float16)c1.y;  o[6]=(_Float16)c1.z;  o[7]=(_Float16)c1.w;
  o[8]=(_Float16)c2.x;  o[9]=(_Float16)c2.y;  o[10]=(_Float16)c2.z; o[11]=(_Float16)c2.w;
  o[12]=(_Float16)c3.x; o[13]=(_Float16)c3.y; o[14]=(_Float16)c3.z; o[15]=(_Float16)c3.w;
  return o;
}

__device__ __forceinline__ v16h load16(const _Float16* __restrict__ p) {
  v8h c0 = *(const v8h*)(p);
  v8h c1 = *(const v8h*)(p + 16);
  v16h o;
#pragma unroll
  for (int j = 0; j < 8; ++j) { o[j] = c0[j]; o[8 + j] = c1[j]; }
  return o;
}

// guarded tail loaders (K % 32 != 0); p already includes row base + off
__device__ __forceinline__ v16h tail32(const float* __restrict__ p, int k, int off, int K) {
  v16h o;
#pragma unroll
  for (int j = 0; j < 8; ++j) {
    int k0 = k + off + j, k1 = k + 16 + off + j;
    o[j]     = (_Float16)((k0 < K) ? p[k + j] : 0.f);
    o[8 + j] = (_Float16)((k1 < K) ? p[k + 16 + j] : 0.f);
  }
  return o;
}

__device__ __forceinline__ v16h tail16(const _Float16* __restrict__ p, int k, int off, int K) {
  v16h o;
#pragma unroll
  for (int j = 0; j < 8; ++j) {
    int k0 = k + off + j, k1 = k + 16 + off + j;
    o[j]     = (k0 < K) ? p[k + j] : (_Float16)0.f;
    o[8 + j] = (k1 < K) ? p[k + 16 + j] : (_Float16)0.f;
  }
  return o;
}

// ----- GEMM kernel -----------------------------------------------------------
// One wave per block, 32x(CT*16) tile. EPI: 0=bias, 1=bias+BN+ELU, 2=ReLU.
// BM: 0 = B is fp32 weight [Nc x K] row-major (A@W^T); 1 = B is f16
// pre-transposed [Nc x K] (used for SpMM with B = h^T in f16).
// RN: divide rows by rowsum(A) (fused adj row-normalization, extra ones-WMMA).
// AC: accumulate into existing C.
template <int CT, int EPI, int BM, bool RN, bool AC>
__global__ __launch_bounds__(32) void wmma_gemm(
    const float* __restrict__ A, int lda,
    const float* __restrict__ Bf, const _Float16* __restrict__ Bh, int ldb,
    const float* __restrict__ bias,
    const float* __restrict__ bng, const float* __restrict__ bnb,
    const float* __restrict__ bnrm, const float* __restrict__ bnrv,
    float* __restrict__ C, int ldc, int M, int Nc, int K) {
  const int lane = threadIdx.x;
  const int hi = lane >> 4;
  const int lo = lane & 15;
  const int off = hi ? 8 : 0;
  const int row0 = blockIdx.x * 32;
  const int col0 = blockIdx.y * (CT * 16);

  // clamped per-lane base pointers -> branch-free main loop, EXEC stays full
  int ra0 = row0 + lo;      if (ra0 >= M) ra0 = M - 1;
  int ra1 = row0 + 16 + lo; if (ra1 >= M) ra1 = M - 1;
  const float* pA0 = A + (size_t)ra0 * lda + off;
  const float* pA1 = A + (size_t)ra1 * lda + off;

  const float*    pBf[CT];
  const _Float16* pBh[CT];
#pragma unroll
  for (int ct = 0; ct < CT; ++ct) {
    int c = col0 + ct * 16 + lo;
    int cc = (c < Nc) ? c : (Nc - 1);
    if (BM == 0) pBf[ct] = Bf + (size_t)cc * ldb + off;
    else         pBh[ct] = Bh + (size_t)cc * ldb + off;
  }

  v8f acc[2][CT];
  v8f rsum[2];
#pragma unroll
  for (int rt = 0; rt < 2; ++rt) {
#pragma unroll
    for (int ct = 0; ct < CT; ++ct)
#pragma unroll
      for (int i = 0; i < 8; ++i) acc[rt][ct][i] = 0.f;
#pragma unroll
    for (int i = 0; i < 8; ++i) rsum[rt][i] = 0.f;
  }

  if (AC) {
#pragma unroll
    for (int rt = 0; rt < 2; ++rt)
#pragma unroll
      for (int ct = 0; ct < CT; ++ct) {
        int ccol = col0 + ct * 16 + lo;
#pragma unroll
        for (int i = 0; i < 8; ++i) {
          int rrow = row0 + rt * 16 + hi * 8 + i;
          acc[rt][ct][i] = (rrow < M && ccol < Nc) ? C[(size_t)rrow * ldc + ccol] : 0.f;
        }
      }
  }

  v16h ones;
#pragma unroll
  for (int j = 0; j < 16; ++j) ones[j] = (_Float16)1.0f;

  constexpr int GS = (CT >= 4) ? 4 : CT;   // B preload group size
  const int kmain = K & ~31;
  if (kmain > 0) {
    v16h a0 = cvt_load32(pA0);
    v16h a1 = cvt_load32(pA1);
    for (int k = 0; k < kmain; k += 32) {
      // double-buffer A: next iteration's tile (clamped -> branch-free)
      const int kn = (k + 32 < kmain) ? (k + 32) : k;
      v16h na0 = cvt_load32(pA0 + kn);
      v16h na1 = cvt_load32(pA1 + kn);
      __builtin_prefetch(pA0 + k + 1024, 0, 1);  // global_prefetch_b8, A stream
      __builtin_prefetch(pA1 + k + 1024, 0, 1);
      if (RN) {
        rsum[0] = __builtin_amdgcn_wmma_f32_16x16x32_f16(false, a0, false, ones,
                                                         (short)0, rsum[0], false, false);
        rsum[1] = __builtin_amdgcn_wmma_f32_16x16x32_f16(false, a1, false, ones,
                                                         (short)0, rsum[1], false, false);
      }
#pragma unroll
      for (int g = 0; g < CT; g += GS) {
        v16h b[GS];
#pragma unroll
        for (int j = 0; j < GS; ++j)
          b[j] = (BM == 0) ? cvt_load32(pBf[g + j] + k) : load16(pBh[g + j] + k);
#pragma unroll
        for (int j = 0; j < GS; ++j) {
          acc[0][g + j] = __builtin_amdgcn_wmma_f32_16x16x32_f16(false, a0, false, b[j],
                                                                 (short)0, acc[0][g + j],
                                                                 false, false);
          acc[1][g + j] = __builtin_amdgcn_wmma_f32_16x16x32_f16(false, a1, false, b[j],
                                                                 (short)0, acc[1][g + j],
                                                                 false, false);
        }
      }
      a0 = na0; a1 = na1;
    }
  }
  if (kmain < K) {  // guarded tail (K % 32 == 16 for K=10000/2000)
    const int k = kmain;
    v16h a0 = tail32(pA0, k, off, K);
    v16h a1 = tail32(pA1, k, off, K);
    if (RN) {
      rsum[0] = __builtin_amdgcn_wmma_f32_16x16x32_f16(false, a0, false, ones,
                                                       (short)0, rsum[0], false, false);
      rsum[1] = __builtin_amdgcn_wmma_f32_16x16x32_f16(false, a1, false, ones,
                                                       (short)0, rsum[1], false, false);
    }
#pragma unroll
    for (int ct = 0; ct < CT; ++ct) {
      v16h b = (BM == 0) ? tail32(pBf[ct], k, off, K) : tail16(pBh[ct], k, off, K);
      acc[0][ct] = __builtin_amdgcn_wmma_f32_16x16x32_f16(false, a0, false, b,
                                                          (short)0, acc[0][ct], false, false);
      acc[1][ct] = __builtin_amdgcn_wmma_f32_16x16x32_f16(false, a1, false, b,
                                                          (short)0, acc[1][ct], false, false);
    }
  }

  // Epilogue: rownorm / bias / BN+ELU / ReLU, guarded stores.
#pragma unroll
  for (int rt = 0; rt < 2; ++rt) {
#pragma unroll
    for (int ct = 0; ct < CT; ++ct) {
      int ccol = col0 + ct * 16 + lo;
      float bv = 0.f, g = 1.f, bb = 0.f, rm = 0.f, rv = 1.f;
      if (ccol < Nc) {
        if (bias) bv = bias[ccol];
        if (EPI == 1) { g = bng[ccol]; bb = bnb[ccol]; rm = bnrm[ccol]; rv = bnrv[ccol]; }
      }
#pragma unroll
      for (int i = 0; i < 8; ++i) {
        float v = acc[rt][ct][i];
        if (RN) v = v / rsum[rt][i];               // every col of rsum tile == row sum
        v += bv;
        if (EPI == 1) {
          v = g * (v - rm) * rsqrtf(rv + 1e-5f) + bb;
          v = (v > 0.f) ? v : (__expf(v) - 1.f);   // ELU
        }
        if (EPI == 2) v = (v > 0.f) ? v : 0.f;     // ReLU
        int rrow = row0 + rt * 16 + hi * 8 + i;
        if (rrow < M && ccol < Nc) C[(size_t)rrow * ldc + ccol] = v;
      }
    }
  }
}

// ----- elementwise helpers ---------------------------------------------------

__global__ void fill0_kernel(float* __restrict__ p, int n) {
  int i = blockIdx.x * blockDim.x + threadIdx.x;
  if (i < n) p[i] = 0.f;
}

// LDS-tiled transpose + f32->f16 convert: dst[c][r] = (f16)src[r][c].
// src: [nrows x 128] fp32; dst: [128 x nrows] f16. block (32,8), 32x32 tiles.
__global__ void transpose_cvt_f16(const float* __restrict__ src, _Float16* __restrict__ dst,
                                  int nrows) {
  __shared__ float tile[32][33];
  const int tx = threadIdx.x, ty = threadIdx.y;
  const int r0 = blockIdx.x * 32;
  const int c0 = blockIdx.y * 32;
#pragma unroll
  for (int i = 0; i < 4; ++i) {
    int r = r0 + ty + 8 * i;
    tile[ty + 8 * i][tx] = (r < nrows) ? src[(size_t)r * 128 + c0 + tx] : 0.f;
  }
  __syncthreads();
#pragma unroll
  for (int i = 0; i < 4; ++i) {
    int c = c0 + ty + 8 * i;     // feature (dst row)
    int r = r0 + tx;             // node (dst col)
    if (r < nrows) dst[(size_t)c * nrows + r] = (_Float16)tile[tx][ty + 8 * i];
  }
}

// F.normalize(h, dim=1): one 128-thread block per row.
__global__ void rownorm128_kernel(const float* __restrict__ src, float* __restrict__ dst) {
  int row = blockIdx.x;
  int t = threadIdx.x;
  float v = src[(size_t)row * 128 + t];
  float ss = v * v;
#pragma unroll
  for (int d = 16; d > 0; d >>= 1) ss += __shfl_xor(ss, d, 32);
  __shared__ float red[4];
  if ((t & 31) == 0) red[t >> 5] = ss;
  __syncthreads();
  float tot = red[0] + red[1] + red[2] + red[3];
  float nrm = fmaxf(sqrtf(tot), 1e-12f);
  dst[(size_t)row * 128 + t] = v / nrm;
}

// z:[N,512] (i,f,g,o), c updated in place, h written to hout.
__global__ void lstm_gates_kernel(const float* __restrict__ z, float* __restrict__ c,
                                  float* __restrict__ hout, int n) {
  int idx = blockIdx.x * blockDim.x + threadIdx.x;
  if (idx >= n) return;
  int row = idx >> 7, col = idx & 127;
  const float* zr = z + (size_t)row * 512;
  float zi = zr[col], zf = zr[128 + col], zg = zr[256 + col], zo = zr[384 + col];
  float ig = 1.f / (1.f + __expf(-zi));
  float fg = 1.f / (1.f + __expf(-zf));
  float gg = tanhf(zg);
  float og = 1.f / (1.f + __expf(-zo));
  float cn = fg * c[idx] + ig * gg;
  c[idx] = cn;
  hout[idx] = og * tanhf(cn);
}

// hcat[:, :128] = elu(bn_in(0.5*(a+b)))  (JK mean, bn_node_in reused)
__global__ void mean_bn_elu_kernel(const float* __restrict__ a, const float* __restrict__ b,
                                   const float* __restrict__ g, const float* __restrict__ bb,
                                   const float* __restrict__ rm, const float* __restrict__ rv,
                                   float* __restrict__ out, int n) {
  int idx = blockIdx.x * blockDim.x + threadIdx.x;
  if (idx >= n) return;
  int row = idx >> 7, col = idx & 127;
  float m = 0.5f * (a[idx] + b[idx]);
  float y = g[col] * (m - rm[col]) * rsqrtf(rv[col] + 1e-5f) + bb[col];
  out[(size_t)row * 192 + col] = (y > 0.f) ? y : (__expf(y) - 1.f);
}

__global__ void log_softmax_kernel(const float* __restrict__ logits, float* __restrict__ out) {
  int row = blockIdx.x;
  int l = threadIdx.x;  // 32 lanes, 20 valid cols
  float v = (l < NOUTC) ? logits[(size_t)row * NOUTC + l] : -INFINITY;
  float mx = v;
#pragma unroll
  for (int d = 16; d > 0; d >>= 1) mx = fmaxf(mx, __shfl_xor(mx, d, 32));
  float e = (l < NOUTC) ? __expf(v - mx) : 0.f;
  float s = e;
#pragma unroll
  for (int d = 16; d > 0; d >>= 1) s += __shfl_xor(s, d, 32);
  if (l < NOUTC) out[(size_t)row * NOUTC + l] = v - mx - __logf(s);
}

// ----- host-side dispatch ----------------------------------------------------

template <int CT, int EPI, bool RN, bool AC>
static void run_gemm_w(const float* A, int lda, const float* B, int ldb,
                       const float* bias, const float* g, const float* bb,
                       const float* rm, const float* rv,
                       float* C, int ldc, int M, int Nc, int K, hipStream_t s) {
  dim3 grid((M + 31) / 32, (Nc + CT * 16 - 1) / (CT * 16));
  hipLaunchKernelGGL(HIP_KERNEL_NAME(wmma_gemm<CT, EPI, 0, RN, AC>),
                     grid, dim3(32), 0, s,
                     A, lda, B, (const _Float16*)nullptr, ldb,
                     bias, g, bb, rm, rv, C, ldc, M, Nc, K);
}

static void run_spmm_rownorm(const float* adj, const _Float16* hT, float* C,
                             int M, int Nc, int K, hipStream_t s) {
  dim3 grid((M + 31) / 32, (Nc + 127) / 128);
  hipLaunchKernelGGL(HIP_KERNEL_NAME(wmma_gemm<8, 0, 1, true, false>),
                     grid, dim3(32), 0, s,
                     adj, K, (const float*)nullptr, hT, K,
                     (const float*)nullptr, (const float*)nullptr, (const float*)nullptr,
                     (const float*)nullptr, (const float*)nullptr, C, Nc, M, Nc, K);
}

static void run_fill0(float* p, int n, hipStream_t s) {
  fill0_kernel<<<(n + 255) / 256, 256, 0, s>>>(p, n);
}

extern "C" void kernel_launch(void* const* d_in, const int* in_sizes, int n_in,
                              void* d_out, int out_size, void* d_ws, size_t ws_size,
                              hipStream_t stream) {
  (void)in_sizes; (void)n_in; (void)out_size; (void)ws_size;
  const int N = NNODES;

  const float* x      = (const float*)d_in[0];
  const float* embed  = (const float*)d_in[1];
  const float* adj    = (const float*)d_in[2];
  const float* W_in   = (const float*)d_in[3];
  const float* b_in   = (const float*)d_in[4];
  const float* bn_in_g  = (const float*)d_in[5];
  const float* bn_in_b  = (const float*)d_in[6];
  const float* bn_in_rm = (const float*)d_in[7];
  const float* bn_in_rv = (const float*)d_in[8];
  const float* W_gs   = (const float*)d_in[9];   // [2,128,256]
  const float* b_gs   = (const float*)d_in[10];  // [2,128]
  const float* Wih0 = (const float*)d_in[11];
  const float* Whh0 = (const float*)d_in[12];
  const float* bih0 = (const float*)d_in[13];
  const float* bhh0 = (const float*)d_in[14];
  const float* Wih1 = (const float*)d_in[15];
  const float* Whh1 = (const float*)d_in[16];
  const float* bih1 = (const float*)d_in[17];
  const float* bhh1 = (const float*)d_in[18];
  const float* W_emb = (const float*)d_in[19];
  const float* b_emb = (const float*)d_in[20];
  const float* bn_emb_g  = (const float*)d_in[21];
  const float* bn_emb_b  = (const float*)d_in[22];
  const float* bn_emb_rm = (const float*)d_in[23];
  const float* bn_emb_rv = (const float*)d_in[24];
  const float* W_fc = (const float*)d_in[25];
  const float* b_fc = (const float*)d_in[26];
  const float* bn_fc_g  = (const float*)d_in[27];
  const float* bn_fc_b  = (const float*)d_in[28];
  const float* bn_fc_rm = (const float*)d_in[29];
  const float* bn_fc_rv = (const float*)d_in[30];
  const float* W_out = (const float*)d_in[31];
  const float* b_out = (const float*)d_in[32];

  // workspace layout (floats)
  float* ws    = (float*)d_ws;
  float* h     = ws;                  // N*128
  float* neigh = h     + 1280000;     // N*128
  float* tbuf  = neigh + 1280000;     // N*128
  float* jk0   = tbuf  + 1280000;     // N*128 (later: hs2[0])
  float* jk1   = jk0   + 1280000;     // N*128 (later: hs2[1])
  float* hs1a  = jk1   + 1280000;     // N*128
  float* hs1b  = hs1a  + 1280000;     // N*128
  float* zbuf  = hs1b  + 1280000;     // N*512
  float* cbuf  = zbuf  + 5120000;     // N*128
  float* hzero = cbuf  + 1280000;     // N*128 (zeros)
  float* hcat  = hzero + 1280000;     // N*192
  float* h2    = hcat  + 1920000;     // N*192
  float* logit = h2    + 1920000;     // N*20
  _Float16* hT = (_Float16*)(logit + 200000);  // 128*N f16 (transposed h)

  const int NE = N * 128;
  const int EB = (NE + 255) / 256;
  const dim3 tgrid((N + 31) / 32, 4), tblk(32, 8);

  // 1. h = elu(bn_in(x @ W_in^T + b_in))          [N,2000]x[128,2000]^T
  run_gemm_w<8, 1, false, false>(x, NFEATC, W_in, NFEATC, b_in,
                                 bn_in_g, bn_in_b, bn_in_rm, bn_in_rv,
                                 h, NHC, N, NHC, NFEATC, stream);

  // 2. Two GraphSAGE layers with fused row-normalized SpMM.
  const float* hin = h;
  float* jks[2] = {jk0, jk1};
  for (int l = 0; l < 2; ++l) {
    // hT = f16(hin)^T, then neigh = (adj @ hin) / rowsum(adj)  (rowsum fused)
    transpose_cvt_f16<<<tgrid, tblk, 0, stream>>>(hin, hT, N);
    run_spmm_rownorm(adj, hT, neigh, N, NHC, N, stream);
    // t = hin @ Wgs[:, :128]^T + b_gs
    run_gemm_w<8, 0, false, false>(hin, NHC, W_gs + (size_t)l * 128 * 256, 256,
                                   b_gs + (size_t)l * 128,
                                   nullptr, nullptr, nullptr, nullptr,
                                   tbuf, NHC, N, NHC, NHC, stream);
    // t = relu(t + neigh @ Wgs[:, 128:]^T)
    run_gemm_w<8, 2, false, true>(neigh, NHC, W_gs + (size_t)l * 128 * 256 + 128, 256,
                                  nullptr, nullptr, nullptr, nullptr, nullptr,
                                  tbuf, NHC, N, NHC, NHC, stream);
    // jk[l] = F.normalize(t, dim=1)
    rownorm128_kernel<<<N, 128, 0, stream>>>(tbuf, jks[l]);
    hin = jks[l];
  }

  // 3. LSTM layer 0 over seq = [jk0, jk1] -> hs1a, hs1b
  run_fill0(hzero, NE, stream);
  run_fill0(cbuf, NE, stream);
  {
    const float* seq[2] = {jk0, jk1};
    float* outs[2] = {hs1a, hs1b};
    const float* hprev = hzero;
    for (int t = 0; t < 2; ++t) {
      run_gemm_w<8, 0, false, false>(seq[t], NHC, Wih0, NHC, bih0,
                                     nullptr, nullptr, nullptr, nullptr,
                                     zbuf, 512, N, 512, NHC, stream);
      run_gemm_w<8, 0, false, true>(hprev, NHC, Whh0, NHC, bhh0,
                                    nullptr, nullptr, nullptr, nullptr,
                                    zbuf, 512, N, 512, NHC, stream);
      lstm_gates_kernel<<<EB, 256, 0, stream>>>(zbuf, cbuf, outs[t], NE);
      hprev = outs[t];
    }
  }

  // 4. LSTM layer 1 over [hs1a, hs1b] -> hs2 (reuse jk0, jk1)
  run_fill0(cbuf, NE, stream);
  {
    const float* seq[2] = {hs1a, hs1b};
    float* outs[2] = {jk0, jk1};
    const float* hprev = hzero;
    for (int t = 0; t < 2; ++t) {
      run_gemm_w<8, 0, false, false>(seq[t], NHC, Wih1, NHC, bih1,
                                     nullptr, nullptr, nullptr, nullptr,
                                     zbuf, 512, N, 512, NHC, stream);
      run_gemm_w<8, 0, false, true>(hprev, NHC, Whh1, NHC, bhh1,
                                    nullptr, nullptr, nullptr, nullptr,
                                    zbuf, 512, N, 512, NHC, stream);
      lstm_gates_kernel<<<EB, 256, 0, stream>>>(zbuf, cbuf, outs[t], NE);
      hprev = outs[t];
    }
  }

  // 5. hcat[:, :128] = elu(bn_in(mean(hs2)))   (JK mean, bn reused)
  mean_bn_elu_kernel<<<EB, 256, 0, stream>>>(jk0, jk1, bn_in_g, bn_in_b,
                                             bn_in_rm, bn_in_rv, hcat, NE);

  // 6. hcat[:, 128:192] = elu(bn_emb(embed @ W_emb^T + b_emb))
  run_gemm_w<4, 1, false, false>(embed, NFEC, W_emb, NFEC, b_emb,
                                 bn_emb_g, bn_emb_b, bn_emb_rm, bn_emb_rv,
                                 hcat + 128, 192, N, NHEC, NFEC, stream);

  // 7. h2 = elu(bn_fc(hcat @ W_fc^T + b_fc))    [N,192]x[192,192]^T
  run_gemm_w<8, 1, false, false>(hcat, 192, W_fc, 192, b_fc,
                                 bn_fc_g, bn_fc_b, bn_fc_rm, bn_fc_rv,
                                 h2, 192, N, 192, 192, stream);

  // 8. logits = h2 @ W_out^T + b_out
  run_gemm_w<2, 0, false, false>(h2, 192, W_out, 192, b_out,
                                 nullptr, nullptr, nullptr, nullptr,
                                 logit, NOUTC, N, NOUTC, 192, stream);

  // 9. log_softmax -> d_out
  log_softmax_kernel<<<N, 32, 0, stream>>>(logit, (float*)d_out);
}
